// LuenbergerStateEstimator_11690900980118
// MI455X (gfx1250) — matrix operand order repaired and checked by reading, not compiled, not run on hardware
//
#include <hip/hip_runtime.h>
#include <cmath>

#define NX   8
#define NU   2
#define NY   2
#define HID  64
#define TSTEPS 2048
#define BATCH  1024

typedef __attribute__((ext_vector_type(16))) __bf16 v16bf;
typedef __attribute__((ext_vector_type(8)))  __bf16 v8bf;
typedef __attribute__((ext_vector_type(8)))  float  v8f;
typedef __attribute__((ext_vector_type(4)))  float  v4f;

#if __has_builtin(__builtin_amdgcn_tanhf)
#define TANHF(x) __builtin_amdgcn_tanhf(x)
#elif __has_builtin(__builtin_amdgcn_tanh_f32)
#define TANHF(x) __builtin_amdgcn_tanh_f32(x)
#else
#define TANHF(x) tanhf(x)
#endif

// One wave (32 lanes) owns 16 batch rows for all 2048 steps.
// A-fragment (16x32 bf16): lane L (L<16) holds row M=L, elems j -> K = (j<8? j : j+8);
//                          lane L (L>=16) holds row M=L-16, elems j -> K = 8 + (j<8? j : j+8).
// B-fragment (32x16 bf16): lane L holds col N=L&15, elem j -> K = 8*(L>>4) + (j<8? j : j+8).
// C/D fragment (16x16 f32): lane L holds col N=L&15, elem r -> row M = r + 8*(L>>4).
__global__ __launch_bounds__(32)
void luenberger_scan_kernel(const float* __restrict__ u,   const float* __restrict__ y,
                            const float* __restrict__ Wf1, const float* __restrict__ bf1,
                            const float* __restrict__ Wf2, const float* __restrict__ bf2,
                            const float* __restrict__ Wg,  const float* __restrict__ bg,
                            const float* __restrict__ Wk1, const float* __restrict__ bk1,
                            const float* __restrict__ Wk2, const float* __restrict__ bk2,
                            float* __restrict__ out)
{
    __shared__ __bf16 hf[16 * HID];   // f-net hidden, row-major [batchrow][hid]
    __shared__ __bf16 hk[16 * HID];   // k-net hidden
    __shared__ float  dxb[16 * NX];   // dx transpose buffer
    __shared__ float  kbuf[16 * 16];  // gain-matrix transpose buffer

    const int lane = threadIdx.x & 31;
    const int half = lane >> 4;
    const int lc   = lane & 15;
    const int row  = blockIdx.x * 16 + lc;   // global batch row this lane owns

    // ---- loop-invariant weight B-fragments (f32 -> bf16, built once) ----
    v16bf Bk1f[4], Bf1f[4], Bf2f[2], Bk2f[2];
    #pragma unroll
    for (int nt = 0; nt < 4; ++nt) {
        #pragma unroll
        for (int j = 0; j < 16; ++j) {
            const int k = 8 * half + (j < 8 ? j : j + 8);
            const float wk = (k < NX)      ? Wk1[k * HID + nt * 16 + lc] : 0.f;
            const float wf = (k < NX + NU) ? Wf1[k * HID + nt * 16 + lc] : 0.f;
            Bk1f[nt][j] = (__bf16)wk;
            Bf1f[nt][j] = (__bf16)wf;
        }
    }
    #pragma unroll
    for (int kc = 0; kc < 2; ++kc) {
        #pragma unroll
        for (int j = 0; j < 16; ++j) {
            const int k = 32 * kc + 8 * half + (j < 8 ? j : j + 8);
            const float wf2 = (lc < NX) ? Wf2[k * NX + lc] : 0.f;
            const float wk2 = Wk2[k * 16 + lc];
            Bf2f[kc][j] = (__bf16)wf2;
            Bk2f[kc][j] = (__bf16)wk2;
        }
    }

    // per-lane bias scalars (folded into C-frag init)
    float bk1v[4], bf1v[4];
    #pragma unroll
    for (int nt = 0; nt < 4; ++nt) { bk1v[nt] = bk1[nt * 16 + lc]; bf1v[nt] = bf1[nt * 16 + lc]; }
    const float bdx = (lc < NX) ? bf2[lc] : 0.f;
    const float bkm = bk2[lc];

    float wg[NX * NY];
    #pragma unroll
    for (int i = 0; i < NX * NY; ++i) wg[i] = Wg[i];
    const float bg0 = bg[0], bg1 = bg[1];

    float x[NX];
    #pragma unroll
    for (int i = 0; i < NX; ++i) x[i] = 0.f;   // lanes >=16 keep x == 0 forever

    const size_t B2 = (size_t)BATCH * 2;

    for (int t = 0; t < TSTEPS; ++t) {
        // hide HBM latency on the serial chain: u/y addresses are state-independent
        if (t + 8 < TSTEPS) {
            __builtin_prefetch(u + (size_t)(t + 8) * B2 + row * 2, 0, 0);
            __builtin_prefetch(y + (size_t)(t + 8) * B2 + row * 2, 0, 0);
        }
        const float2 uv = *(const float2*)(u + (size_t)t * B2 + row * 2);
        const float2 yv = *(const float2*)(y + (size_t)t * B2 + row * 2);

        // ---- lane-local A fragments (K padded to 32) ----
        v16bf ax;
        #pragma unroll
        for (int j = 0; j < 16; ++j) ax[j] = (__bf16)0.f;
        #pragma unroll
        for (int j = 0; j < NX; ++j) ax[j] = (__bf16)x[j];   // lanes>=16 contribute zeros
        v16bf axu = ax;
        // lanes 16..31 hold K=8,9 of the [x;u] row -> the u values
        axu[0] = half ? (__bf16)uv.x : axu[0];
        axu[1] = half ? (__bf16)uv.y : axu[1];

        // ---- layer 1: 4 N-tiles per net, D = A*B + bias ----
        v8f ck[4], cf[4];
        #pragma unroll
        for (int nt = 0; nt < 4; ++nt) {
            #pragma unroll
            for (int r = 0; r < 8; ++r) { ck[nt][r] = bk1v[nt]; cf[nt][r] = bf1v[nt]; }
            ck[nt] = __builtin_amdgcn_wmma_f32_16x16x32_bf16(false, ax,  false, Bk1f[nt],
                                                             (short)0, ck[nt], false, false);
            cf[nt] = __builtin_amdgcn_wmma_f32_16x16x32_bf16(false, axu, false, Bf1f[nt],
                                                             (short)0, cf[nt], false, false);
        }

        // ---- tanh, convert to bf16, stage to LDS row-major [16][64] ----
        #pragma unroll
        for (int nt = 0; nt < 4; ++nt) {
            #pragma unroll
            for (int r = 0; r < 8; ++r) {
                hk[(r + 8 * half) * HID + nt * 16 + lc] = (__bf16)TANHF(ck[nt][r]);
                hf[(r + 8 * half) * HID + nt * 16 + lc] = (__bf16)TANHF(cf[nt][r]);
            }
        }

        // ---- layer 2: gather A-frags (contiguous b128 LDS loads), 2 K-chunks ----
        v8f cdx, ckm;
        #pragma unroll
        for (int r = 0; r < 8; ++r) { cdx[r] = bdx; ckm[r] = bkm; }
        #pragma unroll
        for (int kc = 0; kc < 2; ++kc) {
            const int base = lc * HID + 32 * kc + 8 * half;
            const v8bf flo = *(const v8bf*)&hf[base];
            const v8bf fhi = *(const v8bf*)&hf[base + 16];
            const v8bf klo = *(const v8bf*)&hk[base];
            const v8bf khi = *(const v8bf*)&hk[base + 16];
            v16bf af, ak;
            #pragma unroll
            for (int j = 0; j < 8; ++j) {
                af[j] = flo[j]; af[j + 8] = fhi[j];
                ak[j] = klo[j]; ak[j + 8] = khi[j];
            }
            cdx = __builtin_amdgcn_wmma_f32_16x16x32_bf16(false, af, false, Bf2f[kc],
                                                          (short)0, cdx, false, false);
            ckm = __builtin_amdgcn_wmma_f32_16x16x32_bf16(false, ak, false, Bk2f[kc],
                                                          (short)0, ckm, false, false);
        }

        // ---- transpose D frags back to row-per-lane via LDS ----
        #pragma unroll
        for (int r = 0; r < 8; ++r) kbuf[(r + 8 * half) * 16 + lc] = ckm[r];
        if (lc < NX) {
            #pragma unroll
            for (int r = 0; r < 8; ++r) dxb[(r + 8 * half) * NX + lc] = cdx[r];
        }

        // ---- scalar state update: x += dx + K*(x@Wg + bg - y) (lanes 0..15) ----
        if (!half) {
            const v4f d0 = *(const v4f*)&dxb[lc * NX];
            const v4f d1 = *(const v4f*)&dxb[lc * NX + 4];
            const v4f k0 = *(const v4f*)&kbuf[lc * 16];
            const v4f k1 = *(const v4f*)&kbuf[lc * 16 + 4];
            const v4f k2 = *(const v4f*)&kbuf[lc * 16 + 8];
            const v4f k3 = *(const v4f*)&kbuf[lc * 16 + 12];
            float yh0 = bg0, yh1 = bg1;
            #pragma unroll
            for (int i = 0; i < NX; ++i) { yh0 += x[i] * wg[2 * i]; yh1 += x[i] * wg[2 * i + 1]; }
            const float e0 = yh0 - yv.x, e1 = yh1 - yv.y;
            const float kr[16] = { k0[0],k0[1],k0[2],k0[3], k1[0],k1[1],k1[2],k1[3],
                                   k2[0],k2[1],k2[2],k2[3], k3[0],k3[1],k3[2],k3[3] };
            const float dr[8]  = { d0[0],d0[1],d0[2],d0[3], d1[0],d1[1],d1[2],d1[3] };
            #pragma unroll
            for (int i = 0; i < NX; ++i)
                x[i] = x[i] + dr[i] + kr[2 * i] * e0 + kr[2 * i + 1] * e1;
        }
    }

    if (!half) {
        #pragma unroll
        for (int i = 0; i < NX; ++i) out[(size_t)row * NX + i] = x[i];
    }
}

extern "C" void kernel_launch(void* const* d_in, const int* in_sizes, int n_in,
                              void* d_out, int out_size, void* d_ws, size_t ws_size,
                              hipStream_t stream) {
    (void)in_sizes; (void)n_in; (void)out_size; (void)d_ws; (void)ws_size;
    const float* u   = (const float*)d_in[0];
    const float* y   = (const float*)d_in[1];
    const float* Wf1 = (const float*)d_in[2];
    const float* bf1 = (const float*)d_in[3];
    const float* Wf2 = (const float*)d_in[4];
    const float* bf2 = (const float*)d_in[5];
    const float* Wg  = (const float*)d_in[6];
    const float* bg  = (const float*)d_in[7];
    const float* Wk1 = (const float*)d_in[8];
    const float* bk1 = (const float*)d_in[9];
    const float* Wk2 = (const float*)d_in[10];
    const float* bk2 = (const float*)d_in[11];

    dim3 grid(BATCH / 16);   // 64 independent waves, one 16-row tile each
    dim3 block(32);          // exactly one wave32: no barriers needed anywhere
    luenberger_scan_kernel<<<grid, block, 0, stream>>>(
        u, y, Wf1, bf1, Wf2, bf2, Wg, bg, Wk1, bk1, Wk2, bk2, (float*)d_out);
}
